// SOAPLTBlock_71700184040026
// MI455X (gfx1250) — matrix operand order repaired and checked by reading, not compiled, not run on hardware
//
#include <hip/hip_runtime.h>
#include <hip/hip_bf16.h>

// Sizes (fixed by the reference)
#define B_    4
#define T_    1024
#define D_    1024
#define H_    16
#define DH_   64
#define BT_   4096
#define DFC_  4096

typedef __bf16 bf16_t;
typedef __attribute__((ext_vector_type(16))) __bf16 v16bf;
typedef __attribute__((ext_vector_type(8)))  __bf16 v8bf;
typedef __attribute__((ext_vector_type(8)))  float  v8f;
typedef __attribute__((ext_vector_type(2)))  float  v2f;

union V16U { v16bf v; v8bf h[2]; };

#if defined(__has_builtin)
#if __has_builtin(__builtin_amdgcn_tensor_load_to_lds) && __has_builtin(__builtin_amdgcn_s_wait_tensorcnt)
#define HAVE_TDM 1
#endif
#endif

#ifdef HAVE_TDM
typedef unsigned int tdm_v4u __attribute__((ext_vector_type(4)));
typedef int          tdm_v8i __attribute__((ext_vector_type(8)));
typedef int          tdm_v4i __attribute__((ext_vector_type(4)));

// Issue one TDM DMA: 2-D tile (tile_w x tile_h), bf16 elements, row stride =
// rowlen elements, from global 'gaddr' into LDS byte offset 'lds_addr'.
// D# layout per CDNA5 ISA section 8 (group0 128b, group1 256b, groups 2/3 zero).
__device__ __forceinline__ void tdm_issue_2d(unsigned long long gaddr, unsigned lds_addr,
                                             int rowlen, int nrows, int tile_w, int tile_h) {
  tdm_v4u g0;
  g0.x = 1u;                                              // count=1 (valid), user mode
  g0.y = lds_addr;                                        // lds_addr[31:0]
  g0.z = (unsigned)(gaddr & 0xffffffffu);                 // global_addr[31:0]
  g0.w = (unsigned)((gaddr >> 32) & 0x01ffffffu) | (2u << 30);  // addr[56:32] | type=2
  tdm_v8i g1;
  g1[0] = (int)(1u << 16);                                // data_size=1 (2 bytes/elem)
  g1[1] = (int)(((unsigned)rowlen & 0xffffu) << 16);      // tensor_dim0[15:0]
  g1[2] = (int)((((unsigned)rowlen >> 16) & 0xffffu) |
                (((unsigned)nrows & 0xffffu) << 16));     // dim0[31:16] | dim1[15:0]
  g1[3] = (int)((((unsigned)nrows >> 16) & 0xffffu) |
                (((unsigned)tile_w & 0xffffu) << 16));    // dim1[31:16] | tile_dim0
  g1[4] = (int)((unsigned)tile_h & 0xffffu);              // tile_dim1 (tile_dim2=0)
  g1[5] = rowlen;                                         // tensor_dim0_stride[31:0]
  g1[6] = 0;                                              // stride[47:32] | dim1_stride[15:0]
  g1[7] = 0;
  tdm_v4i z4 = {0, 0, 0, 0};
#if __clang_major__ >= 23
  tdm_v8i z8 = {0, 0, 0, 0, 0, 0, 0, 0};
  __builtin_amdgcn_tensor_load_to_lds(g0, g1, z4, z4, z8, 0);
#else
  __builtin_amdgcn_tensor_load_to_lds(g0, g1, z4, z4, 0);
#endif
}
#endif  // HAVE_TDM

// ---------------------------------------------------------------------------
// f32 -> bf16 conversion (weights)
// ---------------------------------------------------------------------------
__global__ __launch_bounds__(256) void k_f32_to_bf16(const float* __restrict__ in,
                                                     bf16_t* __restrict__ out, int n) {
  for (int i = blockIdx.x * blockDim.x + threadIdx.x; i < n; i += gridDim.x * blockDim.x)
    out[i] = (bf16_t)in[i];
}

// ---------------------------------------------------------------------------
// Row LayerNorm (D=1024) -> bf16, one block (256 thr) per row
// ---------------------------------------------------------------------------
__global__ __launch_bounds__(256) void k_layernorm_bf16(const float* __restrict__ x,
                                                        const float* __restrict__ w,
                                                        bf16_t* __restrict__ out) {
  __shared__ float red[256];
  const int row = blockIdx.x, tid = threadIdx.x;
  const float* xr = x + (size_t)row * D_;
  float v[4]; float s = 0.f;
#pragma unroll
  for (int j = 0; j < 4; j++) { v[j] = xr[tid + 256 * j]; s += v[j]; }
  red[tid] = s; __syncthreads();
  for (int st = 128; st > 0; st >>= 1) { if (tid < st) red[tid] += red[tid + st]; __syncthreads(); }
  const float mu = red[0] * (1.0f / D_); __syncthreads();
  float ss = 0.f;
#pragma unroll
  for (int j = 0; j < 4; j++) { const float d = v[j] - mu; ss += d * d; }
  red[tid] = ss; __syncthreads();
  for (int st = 128; st > 0; st >>= 1) { if (tid < st) red[tid] += red[tid + st]; __syncthreads(); }
  const float rs = rsqrtf(red[0] * (1.0f / D_) + 1e-5f);
#pragma unroll
  for (int j = 0; j < 4; j++) {
    const int i = tid + 256 * j;
    out[(size_t)row * D_ + i] = (bf16_t)((v[j] - mu) * rs * w[i]);
  }
}

// ---------------------------------------------------------------------------
// Tiled bf16 WMMA GEMM:  C[M,N] = A[M,K] @ W[N,K]^T
// block = 256 thr (8 waves); block tile 64(M) x 128(N); wave tile 16 x 64.
// W block (128 x 64K-chunk) staged into LDS by the Tensor Data Mover,
// double-buffered against TENSORcnt; A streamed per-lane from global.
// mode 0: f32 out | mode 1: gelu -> bf16 out | mode 2: scatter into q/k/vt
// ---------------------------------------------------------------------------
__global__ __launch_bounds__(256) void k_gemm_bf16(const bf16_t* __restrict__ A,
                                                   const bf16_t* __restrict__ W,
                                                   float* __restrict__ Cf,
                                                   bf16_t* __restrict__ Cb,
                                                   bf16_t* __restrict__ Qo,
                                                   bf16_t* __restrict__ Ko,
                                                   bf16_t* __restrict__ Vt,
                                                   int M, int N, int K, int mode) {
  const int tid = threadIdx.x;
  const int wv = tid >> 5, ln = tid & 31;
  const int lq = ln & 15, hi = ln >> 4;
  const int mi = wv & 3, ng = wv >> 2;
  const int m0 = blockIdx.y * 64, n0 = blockIdx.x * 128;
  const int arow = m0 + 16 * mi + lq;

  v8f acc[4] = {{}, {}, {}, {}};
  const bf16_t* aptr = A + (size_t)arow * K;

#ifdef HAVE_TDM
  __shared__ bf16_t sW[2][128 * 64];  // 2 x 16 KB double buffer
  const int nstage = K >> 6;          // 64 K-columns per stage
  const unsigned long long wbase = (unsigned long long)(uintptr_t)(W + (size_t)n0 * K);
  const unsigned lds0 = (unsigned)(uintptr_t)(&sW[0][0]);
  const unsigned lds1 = (unsigned)(uintptr_t)(&sW[1][0]);

  if (wv == 0) tdm_issue_2d(wbase, lds0, K, N, 64, 128);
  for (int s = 0; s < nstage; ++s) {
    if (wv == 0) {
      if (s + 1 < nstage) {
        tdm_issue_2d(wbase + (unsigned long long)(s + 1) * 64ull * 2ull,
                     ((s + 1) & 1) ? lds1 : lds0, K, N, 64, 128);
        __builtin_amdgcn_s_wait_tensorcnt(1);  // stage s complete (in-order TDM)
      } else {
        __builtin_amdgcn_s_wait_tensorcnt(0);
      }
    }
    __syncthreads();  // publish stage s
    const bf16_t* wbuf = &sW[s & 1][0];
#pragma unroll
    for (int ks = 0; ks < 2; ++ks) {
      const int kk = s * 64 + ks * 32;
      V16U a;
      a.h[0] = *(const v8bf*)(aptr + kk + 16 * hi);
      a.h[1] = *(const v8bf*)(aptr + kk + 16 * hi + 8);
      if (ks == 0 && kk + 128 < K) __builtin_prefetch((const void*)(aptr + kk + 128), 0, 1);
#pragma unroll
      for (int t = 0; t < 4; t++) {
        const int nl = 16 * (ng * 4 + t) + lq;  // row within 128-row block
        const v16bf b = *(const v16bf*)(wbuf + nl * 64 + ks * 32 + 16 * hi);
        acc[t] = __builtin_amdgcn_wmma_f32_16x16x32_bf16(false, a.v, false, b,
                                                         (short)0, acc[t], false, false);
      }
    }
    __syncthreads();  // all waves done with buffer s&1 before it is re-filled
  }
#else
  for (int kk = 0; kk < K; kk += 32) {
    V16U a;
    a.h[0] = *(const v8bf*)(aptr + kk + 16 * hi);
    a.h[1] = *(const v8bf*)(aptr + kk + 16 * hi + 8);
    if (kk + 128 < K) __builtin_prefetch((const void*)(aptr + kk + 128), 0, 1);
#pragma unroll
    for (int t = 0; t < 4; t++) {
      const int n = n0 + 16 * (ng * 4 + t) + lq;
      const v16bf b = *(const v16bf*)(W + (size_t)n * K + kk + 16 * hi);
      acc[t] = __builtin_amdgcn_wmma_f32_16x16x32_bf16(false, a.v, false, b,
                                                       (short)0, acc[t], false, false);
    }
  }
#endif

#pragma unroll
  for (int t = 0; t < 4; t++) {
    const int ncol = n0 + 16 * (ng * 4 + t) + lq;
#pragma unroll
    for (int j = 0; j < 8; j++) {
      const int mrow = m0 + 16 * mi + j + 8 * hi;
      const float val = acc[t][j];
      if (mode == 0) {
        Cf[(size_t)mrow * N + ncol] = val;
      } else if (mode == 1) {
        const float g = 0.5f * val * (1.0f + erff(val * 0.70710678118f));
        Cb[(size_t)mrow * N + ncol] = (bf16_t)g;
      } else {
        const int bb = mrow >> 10, trow = mrow & 1023;
        const int which = ncol >> 10, rem = ncol & 1023;
        const int hh = rem >> 6, dd = rem & 63;
        const bf16_t bv = (bf16_t)val;
        const size_t bh = (size_t)(bb * H_ + hh);
        if (which == 0)      Qo[(bh * T_ + trow) * DH_ + dd] = bv;
        else if (which == 1) Ko[(bh * T_ + trow) * DH_ + dd] = bv;
        else                 Vt[(bh * DH_ + dd) * T_ + trow] = bv;
      }
    }
  }
}

// ---------------------------------------------------------------------------
// Causal flash attention. One wave per 16-query tile; key tiles of 32.
// S computed transposed (keys=M rows, queries=N cols) so softmax stats are
// per-lane; S^T C-layout feeds P's B-operand with only lane^16 shuffles.
// O accumulated transposed (dh=M, queries=N); V supplied pre-transposed.
// ---------------------------------------------------------------------------
__global__ __launch_bounds__(256) void k_flash_attn(const bf16_t* __restrict__ Q,
                                                    const bf16_t* __restrict__ Kb,
                                                    const bf16_t* __restrict__ Vt,
                                                    bf16_t* __restrict__ O) {
  const int tid = threadIdx.x;
  const int wv = tid >> 5, ln = tid & 31, lq = ln & 15, hi = ln >> 4;
  const int gw = blockIdx.x * 8 + wv;
  const int qt = gw & 63;
  const int hd = (gw >> 6) & 15;
  const int bb = gw >> 10;
  const int q0 = qt * 16;
  const size_t bh = (size_t)(bb * H_ + hd);

  const bf16_t* qbase = Q + (bh * T_ + q0 + lq) * DH_;
  v16bf Qb[2];
  Qb[0] = *(const v16bf*)(qbase + 16 * hi);
  Qb[1] = *(const v16bf*)(qbase + 32 + 16 * hi);

  float m_run = -1e30f, l_run = 0.f;
  v8f Ot[4] = {{}, {}, {}, {}};
  const float scale = 0.125f;  // 1/sqrt(64)

  const int nkt = (q0 + 16 + 31) >> 5;
  for (int kt = 0; kt < nkt; kt++) {
    const int k0 = kt * 32;
    v8f st[2] = {{}, {}};
#pragma unroll
    for (int p = 0; p < 2; p++) {
      const bf16_t* kb = Kb + (bh * T_ + k0 + 16 * p + lq) * DH_;
#pragma unroll
      for (int s = 0; s < 2; s++) {
        V16U a;
        a.h[0] = *(const v8bf*)(kb + 32 * s + 16 * hi);
        a.h[1] = *(const v8bf*)(kb + 32 * s + 16 * hi + 8);
        st[p] = __builtin_amdgcn_wmma_f32_16x16x32_bf16(false, a.v, false, Qb[s],
                                                        (short)0, st[p], false, false);
      }
    }
    const int q = q0 + lq;
    float e1[8], e2[8];
    float tmax = -1e30f;
#pragma unroll
    for (int j = 0; j < 8; j++) {
      const int key1 = k0 + j + 8 * hi;
      const int key2 = key1 + 16;
      e1[j] = (key1 <= q) ? st[0][j] * scale : -1e30f;
      e2[j] = (key2 <= q) ? st[1][j] * scale : -1e30f;
      tmax = fmaxf(tmax, fmaxf(e1[j], e2[j]));
    }
    tmax = fmaxf(tmax, __shfl_xor(tmax, 16, 32));
    const float m_new = fmaxf(m_run, tmax);
    const float alpha = __expf(m_run - m_new);
    float ssum = 0.f;
#pragma unroll
    for (int j = 0; j < 8; j++) {
      e1[j] = __expf(e1[j] - m_new);
      e2[j] = __expf(e2[j] - m_new);
      ssum += e1[j] + e2[j];
    }
    ssum += __shfl_xor(ssum, 16, 32);
    l_run = l_run * alpha + ssum;
    m_run = m_new;
#pragma unroll
    for (int t = 0; t < 4; t++) Ot[t] *= alpha;

    v16bf Pb;
#pragma unroll
    for (int j = 0; j < 8; j++) {
      const float o1 = __shfl_xor(e1[j], 16, 32);
      const float o2 = __shfl_xor(e2[j], 16, 32);
      Pb[j]     = (bf16_t)(hi ? o2 : e1[j]);
      Pb[8 + j] = (bf16_t)(hi ? e2[j] : o1);
    }
#pragma unroll
    for (int t = 0; t < 4; t++) {
      const bf16_t* vb = Vt + (bh * DH_ + 16 * t + lq) * T_ + k0;
      V16U a;
      a.h[0] = *(const v8bf*)(vb + 16 * hi);
      a.h[1] = *(const v8bf*)(vb + 16 * hi + 8);
      Ot[t] = __builtin_amdgcn_wmma_f32_16x16x32_bf16(false, a.v, false, Pb,
                                                      (short)0, Ot[t], false, false);
    }
  }

  const float rl = 1.0f / l_run;
  bf16_t* ob = O + ((size_t)(bb * T_ + q0 + lq)) * D_ + hd * 64;
#pragma unroll
  for (int t = 0; t < 4; t++) {
    v8bf pk;
#pragma unroll
    for (int j = 0; j < 8; j++) pk[j] = (bf16_t)(Ot[t][j] * rl);
    *(v8bf*)(ob + 16 * t + 8 * hi) = pk;
  }
}

// ---------------------------------------------------------------------------
// 64x64 fp32 matmul in LDS via V_WMMA_F32_16X16X4_F32.
// 8 waves x 2 tiles (same M-row pair). Deferred stores let us overwrite inputs.
// ---------------------------------------------------------------------------
__device__ __forceinline__ void mm64_tiles(const float* __restrict__ A,
                                           const float* __restrict__ Bm,
                                           int wv, int lq, int hi, v8f* o0, v8f* o1) {
  v8f a0 = {}, a1 = {};
  const int mi  = (2 * wv) >> 2;
  const int ni0 = (2 * wv) & 3;
  const float* ar = A + (16 * mi + lq) * 64;
#pragma unroll
  for (int k = 0; k < 64; k += 4) {
    const int kk = k + 2 * hi;
    v2f av; av.x = ar[kk]; av.y = ar[kk + 1];
    v2f b0; b0.x = Bm[kk * 64 + 16 * ni0 + lq];       b0.y = Bm[(kk + 1) * 64 + 16 * ni0 + lq];
    v2f b1; b1.x = Bm[kk * 64 + 16 * (ni0 + 1) + lq]; b1.y = Bm[(kk + 1) * 64 + 16 * (ni0 + 1) + lq];
    a0 = __builtin_amdgcn_wmma_f32_16x16x4_f32(false, av, false, b0, (short)0, a0, false, false);
    a1 = __builtin_amdgcn_wmma_f32_16x16x4_f32(false, av, false, b1, (short)0, a1, false, false);
  }
  *o0 = a0; *o1 = a1;
}

__device__ __forceinline__ void mm64_store(float* __restrict__ C, int wv, int lq, int hi,
                                           const v8f& a0, const v8f& a1) {
  const int mi  = (2 * wv) >> 2;
  const int ni0 = (2 * wv) & 3;
#pragma unroll
  for (int j = 0; j < 8; j++) {
    const int row = 16 * mi + j + 8 * hi;
    C[row * 64 + 16 * ni0 + lq]       = a0[j];
    C[row * 64 + 16 * (ni0 + 1) + lq] = a1[j];
  }
}

// ---------------------------------------------------------------------------
// SOAP update (one block = one token):
//   m' = b1*m + (1-b1)*G ; R' = bR*R + (1-bR)*G^T G ; X = (R'+eps I)/||.||_F
//   Newton-Schulz 10x in fp32 WMMA ; U = m'(16x64) @ R'^{-1/2} ; x' = x + g*LN(U)
// ---------------------------------------------------------------------------
__global__ __launch_bounds__(256) void k_soap_update(const float* __restrict__ g,
                                                     const float* __restrict__ m_in,
                                                     const float* __restrict__ R_in,
                                                     const float* __restrict__ x_in,
                                                     const float* __restrict__ ln_w,
                                                     const float* __restrict__ b1p,
                                                     const float* __restrict__ bRp,
                                                     const float* __restrict__ gp,
                                                     float* __restrict__ m_out,
                                                     float* __restrict__ R_out,
                                                     float* __restrict__ x_out) {
  __shared__ float sX[4096], sRn[4096], sP[4096];
  __shared__ float sG[1024], sM[1024];
  __shared__ float red[256];

  const int tid = threadIdx.x;
  const int bt = blockIdx.x;
  const int wv = tid >> 5, ln = tid & 31, lq = ln & 15, hi = ln >> 4;

  const float beta1 = 1.f / (1.f + __expf(-b1p[0]));
  const float betaR = 1.f / (1.f + __expf(-bRp[0]));
  const float graw = gp[0];
  const float gamma = (graw > 20.f) ? graw : log1pf(__expf(graw));  // softplus

  for (int i = tid; i < 1024; i += 256) {
    const float gv = g[(size_t)bt * 1024 + i];
    const float mh = beta1 * m_in[(size_t)bt * 1024 + i] + (1.f - beta1) * gv;
    sG[i] = gv; sM[i] = mh;
    m_out[(size_t)bt * 1024 + i] = mh;
  }
  __syncthreads();

  float ssq = 0.f;
  for (int idx = tid; idx < 4096; idx += 256) {
    const int d = idx >> 6, e = idx & 63;
    float gt = 0.f;
#pragma unroll
    for (int hh = 0; hh < 16; hh++) gt += sG[hh * 64 + d] * sG[hh * 64 + e];
    const float r = betaR * R_in[(size_t)bt * 4096 + idx] + (1.f - betaR) * gt;
    R_out[(size_t)bt * 4096 + idx] = r;
    const float rr = r + ((d == e) ? 1e-6f : 0.f);
    sRn[idx] = rr;
    ssq += rr * rr;
  }
  red[tid] = ssq; __syncthreads();
  for (int st = 128; st > 0; st >>= 1) { if (tid < st) red[tid] += red[tid + st]; __syncthreads(); }
  float norm = sqrtf(red[0]); norm = fmaxf(norm, 1e-6f);
  const float rinv = 1.f / norm;
  __syncthreads();

  for (int idx = tid; idx < 4096; idx += 256) {
    sRn[idx] *= rinv;
    const int d = idx >> 6, e = idx & 63;
    sX[idx] = (d == e) ? 1.f : 0.f;
  }
  __syncthreads();

  for (int it = 0; it < 10; ++it) {
    v8f t0, t1;
    mm64_tiles(sX, sRn, wv, lq, hi, &t0, &t1);   // X @ Rn
    mm64_store(sP, wv, lq, hi, t0, t1);
    __syncthreads();
    mm64_tiles(sP, sX, wv, lq, hi, &t0, &t1);    // (X Rn) @ X
    __syncthreads();                             // all reads of sP done
    mm64_store(sP, wv, lq, hi, t0, t1);          // sP = XRX
    __syncthreads();
    mm64_tiles(sX, sP, wv, lq, hi, &t0, &t1);    // X @ XRX (kept in VGPRs)
    __syncthreads();                             // all reads of sX done
    {
      const int mi = (2 * wv) >> 2, ni0 = (2 * wv) & 3;
#pragma unroll
      for (int j = 0; j < 8; j++) {
        const int row = 16 * mi + j + 8 * hi;
        const int i0 = row * 64 + 16 * ni0 + lq;
        const int i1 = row * 64 + 16 * (ni0 + 1) + lq;
        sX[i0] = 1.5f * sX[i0] - 0.5f * t0[j];
        sX[i1] = 1.5f * sX[i1] - 0.5f * t1[j];
      }
    }
    __syncthreads();
  }

  const float rs = rsqrtf(norm);
  v8f u = {};
  if (wv < 4) {
#pragma unroll
    for (int k = 0; k < 64; k += 4) {
      const int kk = k + 2 * hi;
      v2f av; av.x = sM[lq * 64 + kk];              av.y = sM[lq * 64 + kk + 1];
      v2f bv; bv.x = sX[kk * 64 + 16 * wv + lq];    bv.y = sX[(kk + 1) * 64 + 16 * wv + lq];
      u = __builtin_amdgcn_wmma_f32_16x16x4_f32(false, av, false, bv, (short)0, u, false, false);
    }
  }
  __syncthreads();
  if (wv < 4) {
#pragma unroll
    for (int j = 0; j < 8; j++) sG[(j + 8 * hi) * 64 + 16 * wv + lq] = u[j] * rs;  // sG := U
  }
  __syncthreads();

  float v[4]; float s = 0.f;
#pragma unroll
  for (int j = 0; j < 4; j++) { v[j] = sG[tid + 256 * j]; s += v[j]; }
  red[tid] = s; __syncthreads();
  for (int st = 128; st > 0; st >>= 1) { if (tid < st) red[tid] += red[tid + st]; __syncthreads(); }
  const float mu = red[0] * (1.0f / 1024.f); __syncthreads();
  float ss2 = 0.f;
#pragma unroll
  for (int j = 0; j < 4; j++) { const float d = v[j] - mu; ss2 += d * d; }
  red[tid] = ss2; __syncthreads();
  for (int st = 128; st > 0; st >>= 1) { if (tid < st) red[tid] += red[tid + st]; __syncthreads(); }
  const float rstd = rsqrtf(red[0] * (1.0f / 1024.f) + 1e-5f);
#pragma unroll
  for (int j = 0; j < 4; j++) {
    const int i = tid + 256 * j;
    x_out[(size_t)bt * 1024 + i] =
        x_in[(size_t)bt * 1024 + i] + gamma * (v[j] - mu) * rstd * ln_w[i];
  }
}

// ---------------------------------------------------------------------------
// Host orchestration
// ---------------------------------------------------------------------------
extern "C" void kernel_launch(void* const* d_in, const int* in_sizes, int n_in,
                              void* d_out, int out_size, void* d_ws, size_t ws_size,
                              hipStream_t stream) {
  (void)in_sizes; (void)n_in; (void)out_size; (void)ws_size;
  const float* x     = (const float*)d_in[0];
  const float* m     = (const float*)d_in[1];
  const float* R     = (const float*)d_in[2];
  const float* Wqkv  = (const float*)d_in[3];
  const float* Wo    = (const float*)d_in[4];
  const float* Wfc   = (const float*)d_in[5];
  const float* Wproj = (const float*)d_in[6];
  const float* ln_attn_w    = (const float*)d_in[7];
  const float* ln_mlp_w     = (const float*)d_in[8];
  const float* ln_up_attn_w = (const float*)d_in[9];
  const float* ln_up_mlp_w  = (const float*)d_in[10];
  const float* b1a = (const float*)d_in[11];
  const float* bRa = (const float*)d_in[12];
  const float* ga  = (const float*)d_in[13];
  const float* b1m = (const float*)d_in[14];
  const float* bRm = (const float*)d_in[15];
  const float* gm  = (const float*)d_in[16];

  char* ws = (char*)d_ws;
  size_t off = 0;
  auto alloc = [&](size_t bytes) -> char* {
    char* p = ws + off;
    off = (off + bytes + 255) & ~(size_t)255;
    return p;
  };

  bf16_t* wqkv_b  = (bf16_t*)alloc((size_t)3 * D_ * D_ * 2);
  bf16_t* wo_b    = (bf16_t*)alloc((size_t)D_ * D_ * 2);
  bf16_t* wfc_b   = (bf16_t*)alloc((size_t)DFC_ * D_ * 2);
  bf16_t* wproj_b = (bf16_t*)alloc((size_t)D_ * DFC_ * 2);
  bf16_t* xn_b    = (bf16_t*)alloc((size_t)BT_ * D_ * 2);
  bf16_t* q_b     = (bf16_t*)alloc((size_t)BT_ * D_ * 2);
  bf16_t* k_b     = (bf16_t*)alloc((size_t)BT_ * D_ * 2);
  bf16_t* vt_b    = (bf16_t*)alloc((size_t)BT_ * D_ * 2);
  bf16_t* o_b     = (bf16_t*)alloc((size_t)BT_ * D_ * 2);
  bf16_t* h_b     = (bf16_t*)alloc((size_t)BT_ * DFC_ * 2);
  float*  g_f     = (float*)alloc((size_t)BT_ * D_ * 4);
  float*  m_half  = (float*)alloc((size_t)BT_ * D_ * 4);
  float*  R_half  = (float*)alloc((size_t)BT_ * DH_ * DH_ * 4);
  float*  x_half  = (float*)alloc((size_t)BT_ * D_ * 4);

  float* x_next = (float*)d_out;
  float* m_next = x_next + (size_t)BT_ * D_;
  float* R_next = x_next + (size_t)2 * BT_ * D_;

  // --- weight conversion ---
  k_f32_to_bf16<<<1024, 256, 0, stream>>>(Wqkv,  wqkv_b,  3 * D_ * D_);
  k_f32_to_bf16<<<1024, 256, 0, stream>>>(Wo,    wo_b,    D_ * D_);
  k_f32_to_bf16<<<1024, 256, 0, stream>>>(Wfc,   wfc_b,   DFC_ * D_);
  k_f32_to_bf16<<<1024, 256, 0, stream>>>(Wproj, wproj_b, D_ * DFC_);

  // --- attention substep ---
  k_layernorm_bf16<<<BT_, 256, 0, stream>>>(x, ln_attn_w, xn_b);
  k_gemm_bf16<<<dim3(3 * D_ / 128, BT_ / 64), 256, 0, stream>>>(
      xn_b, wqkv_b, nullptr, nullptr, q_b, k_b, vt_b, BT_, 3 * D_, D_, 2);
  k_flash_attn<<<BT_ / 8, 256, 0, stream>>>(q_b, k_b, vt_b, o_b);
  k_gemm_bf16<<<dim3(D_ / 128, BT_ / 64), 256, 0, stream>>>(
      o_b, wo_b, g_f, nullptr, nullptr, nullptr, nullptr, BT_, D_, D_, 0);
  k_soap_update<<<BT_, 256, 0, stream>>>(g_f, m, R, x, ln_up_attn_w,
                                         b1a, bRa, ga, m_half, R_half, x_half);

  // --- mlp substep ---
  k_layernorm_bf16<<<BT_, 256, 0, stream>>>(x_half, ln_mlp_w, xn_b);
  k_gemm_bf16<<<dim3(DFC_ / 128, BT_ / 64), 256, 0, stream>>>(
      xn_b, wfc_b, nullptr, h_b, nullptr, nullptr, nullptr, BT_, DFC_, D_, 1);
  k_gemm_bf16<<<dim3(D_ / 128, BT_ / 64), 256, 0, stream>>>(
      h_b, wproj_b, g_f, nullptr, nullptr, nullptr, nullptr, BT_, D_, DFC_, 0);
  k_soap_update<<<BT_, 256, 0, stream>>>(g_f, m_half, R_half, x_half, ln_up_mlp_w,
                                         b1m, bRm, gm, m_next, R_next, x_next);
}
